// LongNet_V1_14491219657234
// MI455X (gfx1250) — compile-verified, matrix operand order checked by reading, not run on hardware
//
#include <hip/hip_runtime.h>
#include <hip/hip_bf16.h>
#include <math.h>

// ---------------------------------------------------------------------------
// Fully-fused MLP (11-20-20-30-30-40-40-40-20-10-1, swish x9 + relu) on
// gfx1250 using v_wmma_f32_16x16x32_f16. One 16-row tile per wave32.
// Weights are pre-swizzled once per block into the ISA B-fragment layout in
// LDS; activations bounce through a per-wave LDS row buffer between layers.
// Swish uses native v_exp_f32 + v_rcp_f32 (TRANS pipe, co-executes w/ WMMA).
// ---------------------------------------------------------------------------

typedef _Float16 h8   __attribute__((ext_vector_type(8)));
typedef _Float16 v16h __attribute__((ext_vector_type(16)));
typedef float    v8f  __attribute__((ext_vector_type(8)));

#define NLAYER 10
#define WAVES  8          // waves per block (256 threads)
#define ACTW   64         // padded activation row width (covers K up to 64)

__device__ __forceinline__ v16h combine16(h8 lo, h8 hi) {
    v16h r;
#pragma unroll
    for (int i = 0; i < 8; ++i) { r[i] = lo[i]; r[i + 8] = hi[i]; }
    return r;
}

// swish(v) = v * sigmoid(v), with native exp2 / rcp (no IEEE div expansion).
__device__ __forceinline__ float fast_swish(float v) {
    const float e = __builtin_amdgcn_exp2f(v * -1.44269504088896f);
    return v * __builtin_amdgcn_rcpf(1.0f + e);
}

struct MlpParams {
    const float* x;
    float*       out;
    const float* W[NLAYER];
    const float* B[NLAYER];
    long long    nrows;
    long long    ntiles;
};

// One hidden layer: act = swish(A(16xK) * B(KxN) + bias).
// N-tiles entirely beyond DOUT are pure padding: skip WMMA, store zeros.
// (nt loop is fully unrolled; the `if` folds at compile time per iteration.)
template <int KT, int NT, int DOUT>
__device__ __forceinline__ void run_layer(_Float16* actW, const _Float16* wb,
                                          const float* bias, int lane) {
    const int m  = lane & 15;
    const int b0 = (lane & 16) ? 8 : 0;     // lane-half K base per ISA layout
    const int Mb = (lane & 16) ? 8 : 0;
    const _Float16* arow = actW + m * ACTW + b0;

    v16h a[KT];
#pragma unroll
    for (int ks = 0; ks < KT; ++ks) {
        h8 lo = *(const h8*)(arow + ks * 32);
        h8 hi = *(const h8*)(arow + ks * 32 + 16);
        a[ks] = combine16(lo, hi);
    }
#pragma unroll
    for (int nt = 0; nt < NT; ++nt) {
        _Float16* ocol = actW + nt * 16 + m;          // column (lane owns N=m)
        if (DOUT <= nt * 16) {
            // fully padded output tile: result is exactly zero
#pragma unroll
            for (int r = 0; r < 8; ++r) ocol[(Mb + r) * ACTW] = (_Float16)0.f;
        } else {
            v8f c = {0.f, 0.f, 0.f, 0.f, 0.f, 0.f, 0.f, 0.f};
#pragma unroll
            for (int ks = 0; ks < KT; ++ks) {
                const _Float16* bp = wb + (ks * NT + nt) * 512 + lane * 16;
                v16h b = combine16(*(const h8*)bp, *(const h8*)(bp + 8));
                c = __builtin_amdgcn_wmma_f32_16x16x32_f16(
                        false, a[ks], false, b, (short)0, c, false, false);
            }
            const float bv = bias[nt * 16 + m];
#pragma unroll
            for (int r = 0; r < 8; ++r) {
                ocol[(Mb + r) * ACTW] = (_Float16)fast_swish(c[r] + bv);
            }
        }
    }
}

// Final layer (10 -> 1) + ReLU, writes 16 contiguous fp32 outputs.
__device__ __forceinline__ void run_final(const _Float16* actW, const _Float16* wb,
                                          const float* bias, int lane,
                                          float* outp, long long rowBase,
                                          long long nrows) {
    const int m  = lane & 15;
    const int b0 = (lane & 16) ? 8 : 0;
    const _Float16* arow = actW + m * ACTW + b0;
    v16h a = combine16(*(const h8*)arow, *(const h8*)(arow + 16));
    const _Float16* bp = wb + lane * 16;
    v16h b = combine16(*(const h8*)bp, *(const h8*)(bp + 8));
    v8f c = {0.f, 0.f, 0.f, 0.f, 0.f, 0.f, 0.f, 0.f};
    c = __builtin_amdgcn_wmma_f32_16x16x32_f16(
            false, a, false, b, (short)0, c, false, false);
    if (m == 0) {
        const float bv = bias[0];
        const int Mb = (lane & 16) ? 8 : 0;
        if (rowBase + 16 <= nrows) {              // common, uniform fast path
#pragma unroll
            for (int r = 0; r < 8; ++r)
                outp[Mb + r] = fmaxf(c[r] + bv, 0.f);
        } else {
#pragma unroll
            for (int r = 0; r < 8; ++r)
                if (rowBase + Mb + r < nrows)
                    outp[Mb + r] = fmaxf(c[r] + bv, 0.f);
        }
    }
}

__global__ __launch_bounds__(WAVES * 32)
void mlp_fused_kernel(MlpParams p) {
    // Layer config (compile-time tables; all uses fully unrolled).
    const int DIN[NLAYER]  = {11, 20, 20, 30, 30, 40, 40, 40, 20, 10};
    const int DOUT[NLAYER] = {20, 20, 30, 30, 40, 40, 40, 20, 10, 1};
    const int KT[NLAYER]   = {1, 1, 1, 1, 1, 2, 2, 2, 1, 1};
    const int NT[NLAYER]   = {2, 2, 2, 2, 4, 4, 4, 2, 2, 1};
    const int FB[NLAYER]   = {0, 1024, 2048, 3072, 4096, 6144, 10240, 14336, 16384, 17408};
    const int BB[NLAYER]   = {0, 32, 64, 96, 128, 192, 256, 320, 352, 384};

    __shared__ __align__(32) _Float16 wB[17920];          // swizzled B frags
    __shared__ __align__(32) _Float16 act[WAVES * 16 * ACTW];
    __shared__ float bL[400];                             // padded biases

    // ---- one-time per-block: fp32 weights -> f16 B-fragment layout ----
    for (int l = 0; l < NLAYER; ++l) {
        const int din = DIN[l], dout = DOUT[l], nt = NT[l];
        const int nHalf = KT[l] * nt * 512;
        const float* W = p.W[l];
        for (int idx = threadIdx.x; idx < nHalf; idx += blockDim.x) {
            const int frag = idx >> 9, pos = idx & 511;
            const int ln = pos >> 4, i = pos & 15;
            const int ks = frag / nt, ntile = frag - ks * nt;
            const int kb = (ln & 16) ? 8 : 0;
            const int k  = ks * 32 + ((i < 8) ? (kb + i) : (16 + kb + (i - 8)));
            const int n  = ntile * 16 + (ln & 15);
            const float v = (k < din && n < dout) ? W[n * din + k] : 0.f;
            wB[FB[l] + idx] = (_Float16)v;
        }
        const float* Bv = p.B[l];
        for (int idx = threadIdx.x; idx < nt * 16; idx += blockDim.x)
            bL[BB[l] + idx] = (idx < dout) ? Bv[idx] : 0.f;
    }
    __syncthreads();

    const int wave = threadIdx.x >> 5;
    const int lane = threadIdx.x & 31;
    _Float16* actW = act + wave * (16 * ACTW);

    // Zero this wave's activation buffer ONCE. After that, any stale data in
    // padded K columns is a finite f16 value multiplied by zero B rows -> 0,
    // so per-tile re-zeroing is unnecessary.
    {
        h8 z = {};
#pragma unroll
        for (int t = 0; t < 4; ++t)
            ((h8*)actW)[lane + t * 32] = z;              // 16*64 halfs total
    }

    const long long stride = (long long)gridDim.x * WAVES;
    for (long long tile = (long long)blockIdx.x * WAVES + wave;
         tile < p.ntiles; tile += stride) {
        const long long rowBase = tile * 16;

        // Load 16 rows x 11 features (176 contiguous floats) -> f16 in LDS.
        {
            const float* xsrc = p.x + rowBase * 11;
            const long long navail = p.nrows * 11 - rowBase * 11;
#pragma unroll
            for (int t = 0; t < 6; ++t) {
                const int idx = lane + t * 32;
                if (idx < 176 && (long long)idx < navail) {
                    const float v = xsrc[idx];
                    const int r = idx / 11;               // const-div -> mulhi
                    const int c = idx - r * 11;
                    actW[r * ACTW + c] = (_Float16)v;
                }
            }
        }

        run_layer<1, 2, 20>(actW, wB + FB[0], bL + BB[0], lane);
        run_layer<1, 2, 20>(actW, wB + FB[1], bL + BB[1], lane);
        run_layer<1, 2, 30>(actW, wB + FB[2], bL + BB[2], lane);
        run_layer<1, 2, 30>(actW, wB + FB[3], bL + BB[3], lane);
        run_layer<1, 4, 40>(actW, wB + FB[4], bL + BB[4], lane);
        run_layer<2, 4, 40>(actW, wB + FB[5], bL + BB[5], lane);
        run_layer<2, 4, 40>(actW, wB + FB[6], bL + BB[6], lane);
        run_layer<2, 2, 20>(actW, wB + FB[7], bL + BB[7], lane);
        run_layer<1, 2, 10>(actW, wB + FB[8], bL + BB[8], lane);  // nt=1 tile skipped
        run_final(actW, wB + FB[9], bL + BB[9], lane,
                  p.out + rowBase, rowBase, p.nrows);
    }
}

extern "C" void kernel_launch(void* const* d_in, const int* in_sizes, int n_in,
                              void* d_out, int out_size, void* d_ws, size_t ws_size,
                              hipStream_t stream) {
    (void)n_in; (void)d_ws; (void)ws_size; (void)out_size;
    MlpParams p;
    p.x   = (const float*)d_in[0];
    p.out = (float*)d_out;
    for (int l = 0; l < NLAYER; ++l) {
        p.W[l] = (const float*)d_in[1 + 2 * l];
        p.B[l] = (const float*)d_in[2 + 2 * l];
    }
    p.nrows  = (long long)(in_sizes[0] / 11);
    p.ntiles = (p.nrows + 15) / 16;        // 2,000,000 / 16 = 125,000

    const int grid = 1024;                 // 8192 waves, ~15 tiles each
    hipLaunchKernelGGL(mlp_fused_kernel, dim3(grid), dim3(WAVES * 32), 0, stream, p);
}
